// StiefelGovernor_64759516889555
// MI455X (gfx1250) — matrix operand order repaired and checked by reading, not compiled, not run on hardware
//
#include <hip/hip_runtime.h>

typedef __attribute__((ext_vector_type(16))) __bf16 v16bf;
typedef __attribute__((ext_vector_type(8)))  float  v8f;

union Frag { v16bf v; uint4 q[2]; };

#define B_ROWS   16384
#define K_DIM    1024
#define N_DIM    768
#define TM       128
#define TN       128
#define TK       32
#define KST      40   // LDS row stride in bf16 elements (80 B: 16B-aligned, bank-staggered)
#define NKCHUNK  (K_DIM / TK)   // 32

static __device__ __forceinline__ v8f wmma_bf16(v16bf a, v16bf b, v8f c) {
    return __builtin_amdgcn_wmma_f32_16x16x32_bf16(
        false, a, false, b, (short)0, c, false, false);
}

static __device__ __forceinline__ unsigned lds_addr32(const void* p) {
    // generic LDS pointer: addr[31:0] is the LDS byte offset (flat aperture rule)
    return (unsigned)(uintptr_t)p;
}

// ---------------------------------------------------------------------------
// Prepass: split f32 -> bf16 hi + bf16 lo (x ~= hi + lo), 8 elements/thread.
// ---------------------------------------------------------------------------
__global__ __launch_bounds__(256) void
split_f32_to_bf16x2(const float* __restrict__ src, unsigned short* __restrict__ hi,
                    unsigned short* __restrict__ lo, int n8) {
    const int i = blockIdx.x * 256 + threadIdx.x;
    if (i >= n8) return;
    float f[8];
    *(float4*)(f)     = *(const float4*)(src + (size_t)i * 8);
    *(float4*)(f + 4) = *(const float4*)(src + (size_t)i * 8 + 4);
    union { __bf16 b[8]; uint4 u; } H, L;
    #pragma unroll
    for (int j = 0; j < 8; ++j) {
        const __bf16 h = (__bf16)f[j];
        H.b[j] = h;
        L.b[j] = (__bf16)(f[j] - (float)h);
    }
    *(uint4*)(hi + (size_t)i * 8) = H.u;
    *(uint4*)(lo + (size_t)i * 8) = L.u;
}

// ---------------------------------------------------------------------------
// Fast GEMM: Z = X @ W^T with pre-split bf16 hi/lo operands.
// Double-buffered LDS staging via global_load_async_to_lds_b128 (ASYNCcnt).
// 256 threads / 8 waves, 128x128 tile, wave grid 2(M) x 4(N), 4x2 accums.
// ---------------------------------------------------------------------------
__global__ __launch_bounds__(256) void
stiefel_gemm_async(const unsigned short* __restrict__ Xh, const unsigned short* __restrict__ Xl,
                   const unsigned short* __restrict__ Wh, const unsigned short* __restrict__ Wl,
                   float* __restrict__ Z) {
    // [buf][tile: 0=Ah 1=Al 2=Bh 3=Bl][row*KST + k]
    __shared__ __bf16 lds[2][4][TM * KST];   // 80 KB

    const int bn   = blockIdx.x % (N_DIM / TN);
    const int bm   = blockIdx.x / (N_DIM / TN);
    const int t    = threadIdx.x;
    const int lane = t & 31;
    const int wv   = t >> 5;
    const int wm   = wv >> 2;
    const int wn   = wv & 3;
    const int half = lane >> 4;
    const int l16  = lane & 15;

    const unsigned short* srcs[4] = { Xh, Xl, Wh, Wl };
    const int             rbase[4] = { bm * TM, bm * TM, bn * TN, bn * TN };

    // Issue one K-chunk's staging: 4 tiles * 512 16B-chunks / 256 thr = 8 asyncs.
    auto issue = [&](int buf, int k0) {
        #pragma unroll
        for (int i = 0; i < 2; ++i) {
            const int idx = t + i * 256;       // 0..511
            const int row = idx >> 2;          // 0..127
            const int c   = (idx & 3) * 8;     // bf16 offset in row: 0,8,16,24
            #pragma unroll
            for (int tl = 0; tl < 4; ++tl) {
                const unsigned short* g =
                    srcs[tl] + (size_t)(rbase[tl] + row) * K_DIM + k0 + c;
                const unsigned l = lds_addr32(&lds[buf][tl][row * KST + c]);
                asm volatile("global_load_async_to_lds_b128 %0, %1, off"
                             :: "v"(l), "v"(g) : "memory");
            }
        }
    };

    v8f acc[4][2];
    #pragma unroll
    for (int mt = 0; mt < 4; ++mt)
        #pragma unroll
        for (int ct = 0; ct < 2; ++ct)
            acc[mt][ct] = (v8f){0.f, 0.f, 0.f, 0.f, 0.f, 0.f, 0.f, 0.f};

    issue(0, 0);
    for (int kc = 0; kc < NKCHUNK; ++kc) {
        const int cur = kc & 1;
        if (kc + 1 < NKCHUNK) {
            issue(cur ^ 1, (kc + 1) * TK);
            asm volatile("s_wait_asynccnt 0x8" ::: "memory");  // chunk kc landed
        } else {
            asm volatile("s_wait_asynccnt 0x0" ::: "memory");
        }
        __syncthreads();   // all waves' async writes for chunk kc visible

        const __bf16* Ah = &lds[cur][0][0];
        const __bf16* Al = &lds[cur][1][0];
        const __bf16* Bh = &lds[cur][2][0];
        const __bf16* Bl = &lds[cur][3][0];

        // A fragments: 16-bit A 16x32 layout (lane halves pick K sub-ranges).
        Frag ah[4], al[4];
        const int aklo = half * 8;
        #pragma unroll
        for (int mt = 0; mt < 4; ++mt) {
            const int r = (wm * 64 + mt * 16 + l16) * KST;
            ah[mt].q[0] = *(const uint4*)&Ah[r + aklo];
            ah[mt].q[1] = *(const uint4*)&Ah[r + 16 + aklo];
            al[mt].q[0] = *(const uint4*)&Al[r + aklo];
            al[mt].q[1] = *(const uint4*)&Al[r + 16 + aklo];
        }
        // B fragments (B 32x16: same K order as A: K 8h..8h+7, 16+8h..16+8h+7) + WMMA.
        const int bkb = half * 8;
        #pragma unroll
        for (int ct = 0; ct < 2; ++ct) {
            Frag bh, bl;
            const int r = (wn * 32 + ct * 16 + l16) * KST;
            bh.q[0] = *(const uint4*)&Bh[r + bkb];
            bh.q[1] = *(const uint4*)&Bh[r + 16 + bkb];
            bl.q[0] = *(const uint4*)&Bl[r + bkb];
            bl.q[1] = *(const uint4*)&Bl[r + 16 + bkb];
            #pragma unroll
            for (int mt = 0; mt < 4; ++mt) {
                acc[mt][ct] = wmma_bf16(ah[mt].v, bh.v, acc[mt][ct]); // hi*hi
                acc[mt][ct] = wmma_bf16(ah[mt].v, bl.v, acc[mt][ct]); // hi*lo
                acc[mt][ct] = wmma_bf16(al[mt].v, bh.v, acc[mt][ct]); // lo*hi
            }
        }
        __syncthreads();   // reads done before next chunk's asyncs overwrite
    }

    const int mrow0 = bm * TM + wm * 64;
    const int ncol0 = bn * TN + wn * 32 + l16;
    #pragma unroll
    for (int mt = 0; mt < 4; ++mt)
        #pragma unroll
        for (int ct = 0; ct < 2; ++ct)
            #pragma unroll
            for (int r = 0; r < 8; ++r)
                Z[(size_t)(mrow0 + mt * 16 + r + half * 8) * N_DIM + ncol0 + ct * 16] =
                    acc[mt][ct][r];
}

// ---------------------------------------------------------------------------
// Fallback GEMM (split done in-loop) when workspace is too small.
// ---------------------------------------------------------------------------
__global__ __launch_bounds__(256) void
stiefel_gemm_bf16x3(const float* __restrict__ X, const float* __restrict__ W,
                    float* __restrict__ Z) {
    __shared__ __bf16 Ah[TM * KST];
    __shared__ __bf16 Al[TM * KST];
    __shared__ __bf16 Bh[TN * KST];
    __shared__ __bf16 Bl[TN * KST];

    const int bn   = blockIdx.x % (N_DIM / TN);
    const int bm   = blockIdx.x / (N_DIM / TN);
    const int t    = threadIdx.x;
    const int lane = t & 31;
    const int wv   = t >> 5;
    const int wm   = wv >> 2;
    const int wn   = wv & 3;
    const int half = lane >> 4;
    const int l16  = lane & 15;

    v8f acc[4][2];
    #pragma unroll
    for (int mt = 0; mt < 4; ++mt)
        #pragma unroll
        for (int ct = 0; ct < 2; ++ct)
            acc[mt][ct] = (v8f){0.f, 0.f, 0.f, 0.f, 0.f, 0.f, 0.f, 0.f};

    for (int k0 = 0; k0 < K_DIM; k0 += TK) {
        __syncthreads();
        #pragma unroll
        for (int i = 0; i < 4; ++i) {
            const int idx = t + i * 256;
            const int row = idx >> 3;
            const int kq  = (idx & 7) * 4;
            const float4 xa = *(const float4*)(X + (size_t)(bm * TM + row) * K_DIM + k0 + kq);
            const float4 wa = *(const float4*)(W + (size_t)(bn * TN + row) * K_DIM + k0 + kq);
            union { __bf16 b[4]; uint2 u; } xh, xl, wh, wl;
            #pragma unroll
            for (int j = 0; j < 4; ++j) {
                const float xv = ((const float*)&xa)[j];
                const __bf16 h = (__bf16)xv;
                xh.b[j] = h; xl.b[j] = (__bf16)(xv - (float)h);
                const float wf = ((const float*)&wa)[j];
                const __bf16 g = (__bf16)wf;
                wh.b[j] = g; wl.b[j] = (__bf16)(wf - (float)g);
            }
            *(uint2*)&Ah[row * KST + kq] = xh.u;
            *(uint2*)&Al[row * KST + kq] = xl.u;
            *(uint2*)&Bh[row * KST + kq] = wh.u;
            *(uint2*)&Bl[row * KST + kq] = wl.u;
        }
        __syncthreads();

        Frag ah[4], al[4];
        const int aklo = half * 8;
        #pragma unroll
        for (int mt = 0; mt < 4; ++mt) {
            const int r = (wm * 64 + mt * 16 + l16) * KST;
            ah[mt].q[0] = *(const uint4*)&Ah[r + aklo];
            ah[mt].q[1] = *(const uint4*)&Ah[r + 16 + aklo];
            al[mt].q[0] = *(const uint4*)&Al[r + aklo];
            al[mt].q[1] = *(const uint4*)&Al[r + 16 + aklo];
        }
        const int bkb = half * 8;
        #pragma unroll
        for (int ct = 0; ct < 2; ++ct) {
            Frag bh, bl;
            const int r = (wn * 32 + ct * 16 + l16) * KST;
            bh.q[0] = *(const uint4*)&Bh[r + bkb];
            bh.q[1] = *(const uint4*)&Bh[r + 16 + bkb];
            bl.q[0] = *(const uint4*)&Bl[r + bkb];
            bl.q[1] = *(const uint4*)&Bl[r + 16 + bkb];
            #pragma unroll
            for (int mt = 0; mt < 4; ++mt) {
                acc[mt][ct] = wmma_bf16(ah[mt].v, bh.v, acc[mt][ct]);
                acc[mt][ct] = wmma_bf16(ah[mt].v, bl.v, acc[mt][ct]);
                acc[mt][ct] = wmma_bf16(al[mt].v, bh.v, acc[mt][ct]);
            }
        }
    }

    const int mrow0 = bm * TM + wm * 64;
    const int ncol0 = bn * TN + wn * 32 + l16;
    #pragma unroll
    for (int mt = 0; mt < 4; ++mt)
        #pragma unroll
        for (int ct = 0; ct < 2; ++ct)
            #pragma unroll
            for (int r = 0; r < 8; ++r)
                Z[(size_t)(mrow0 + mt * 16 + r + half * 8) * N_DIM + ncol0 + ct * 16] =
                    acc[mt][ct][r];
}

// ---------------------------------------------------------------------------
// In-place Stiefel retraction: one wave32 per row, Q = M * chol(M^T M)^{-1}.
// ---------------------------------------------------------------------------
__global__ __launch_bounds__(256) void
stiefel_qr_rows(float* __restrict__ Z) {
    const int wave = (int)((blockIdx.x * blockDim.x + threadIdx.x) >> 5);
    const int lane = threadIdx.x & 31;
    if (wave >= B_ROWS) return;
    float* rowp = Z + (size_t)wave * N_DIM;

    float f[24];
    #pragma unroll
    for (int i = 0; i < 6; ++i)
        *(float4*)(f + 4 * i) = *(const float4*)(rowp + lane * 24 + 4 * i);

    float s00 = 0.f, s01 = 0.f, s02 = 0.f, s11 = 0.f, s12 = 0.f, s22 = 0.f;
    #pragma unroll
    for (int r = 0; r < 8; ++r) {
        const float a = f[3 * r], b = f[3 * r + 1], c = f[3 * r + 2];
        s00 += a * a; s01 += a * b; s02 += a * c;
        s11 += b * b; s12 += b * c; s22 += c * c;
    }
    #pragma unroll
    for (int m = 1; m < 32; m <<= 1) {
        s00 += __shfl_xor(s00, m, 32);
        s01 += __shfl_xor(s01, m, 32);
        s02 += __shfl_xor(s02, m, 32);
        s11 += __shfl_xor(s11, m, 32);
        s12 += __shfl_xor(s12, m, 32);
        s22 += __shfl_xor(s22, m, 32);
    }
    const float r00 = sqrtf(s00),              i00 = 1.f / r00;
    const float r01 = s01 * i00;
    const float r02 = s02 * i00;
    const float r11 = sqrtf(s11 - r01 * r01),  i11 = 1.f / r11;
    const float r12 = (s12 - r01 * r02) * i11;
    const float r22 = sqrtf(s22 - r02 * r02 - r12 * r12), i22 = 1.f / r22;

    #pragma unroll
    for (int r = 0; r < 8; ++r) {
        const float q0 = f[3 * r] * i00;
        const float q1 = (f[3 * r + 1] - r01 * q0) * i11;
        const float q2 = (f[3 * r + 2] - r02 * q0 - r12 * q1) * i22;
        f[3 * r] = q0; f[3 * r + 1] = q1; f[3 * r + 2] = q2;
    }
    #pragma unroll
    for (int i = 0; i < 6; ++i)
        *(float4*)(rowp + lane * 24 + 4 * i) = *(const float4*)(f + 4 * i);
}

extern "C" void kernel_launch(void* const* d_in, const int* in_sizes, int n_in,
                              void* d_out, int out_size, void* d_ws, size_t ws_size,
                              hipStream_t stream) {
    (void)in_sizes; (void)n_in; (void)out_size;
    const float* X = (const float*)d_in[0];   // (16384, 1024)
    const float* W = (const float*)d_in[1];   // (768, 1024)
    float* Z = (float*)d_out;                 // (16384, 768)

    const size_t nX = (size_t)B_ROWS * K_DIM;        // 16,777,216
    const size_t nW = (size_t)N_DIM * K_DIM;         //    786,432
    const size_t need = (nX + nW) * 2 * 2;           // hi+lo, 2 B each

    if (ws_size >= need && d_ws != nullptr) {
        unsigned short* xh = (unsigned short*)d_ws;
        unsigned short* xl = xh + nX;
        unsigned short* wh = xl + nX;
        unsigned short* wl = wh + nW;
        split_f32_to_bf16x2<<<dim3((unsigned)(nX / 8 / 256)), dim3(256), 0, stream>>>(X, xh, xl, (int)(nX / 8));
        split_f32_to_bf16x2<<<dim3((unsigned)(nW / 8 / 256)), dim3(256), 0, stream>>>(W, wh, wl, (int)(nW / 8));
        stiefel_gemm_async<<<dim3((B_ROWS / TM) * (N_DIM / TN)), dim3(256), 0, stream>>>(xh, xl, wh, wl, Z);
    } else {
        stiefel_gemm_bf16x3<<<dim3((B_ROWS / TM) * (N_DIM / TN)), dim3(256), 0, stream>>>(X, W, Z);
    }
    stiefel_qr_rows<<<dim3(B_ROWS / 8), dim3(256), 0, stream>>>(Z);
}
